// CodeformerLM_85392539779417
// MI455X (gfx1250) — matrix operand order repaired and checked by reading, not compiled, not run on hardware
//
#include <hip/hip_runtime.h>
#include <hip/hip_bf16.h>

// ---------------------------------------------------------------------------
// CodeformerLM forward for MI455X (gfx1250, wave32, WMMA).
// Heavy GEMMs run through V_WMMA_F32_16X16X32_BF16 (fp32 accumulate).
// Weights converted fp32 -> bf16, transposed to [N][K] once per launch so
// B fragments are contiguous 32B loads in the same K order as A.
// Each wave owns a 16x64 output strip: one A fragment feeds 4 WMMAs.
// ---------------------------------------------------------------------------

typedef __attribute__((ext_vector_type(16))) __bf16 bf16x16;
typedef __attribute__((ext_vector_type(8)))  float  floatx8;

constexpr int Bn  = 2;
constexpr int Cn  = 16;
constexpr int Tn  = 64;
constexpr int Hn  = 512;
constexpr int Fn  = 2048;
constexpr int Ln  = 2;
constexpr int Vn  = 32000;
constexpr int NH  = 8;
constexpr int DH  = 64;       // Hn / NH
constexpr int S2n = Cn + Tn;  // 80
constexpr float NEGV = -1.0e9f;

__device__ __forceinline__ float gelu_tanh(float x) {
  // jax.nn.gelu default (approximate=True, tanh form)
  return 0.5f * x * (1.0f + tanhf(0.7978845608028654f * (x + 0.044715f * x * x * x)));
}

// ---------------------------------------------------------------------------
// Weight convert: fp32 [K][N] row-major  ->  bf16 transposed [N][K]
// ---------------------------------------------------------------------------
__global__ void wconv_kernel(const float* __restrict__ src,
                             unsigned short* __restrict__ dst, int K, int N) {
  size_t i = (size_t)blockIdx.x * blockDim.x + threadIdx.x;
  size_t total = (size_t)K * N;
  if (i >= total) return;
  int k = (int)(i / N);
  int n = (int)(i % N);
  __bf16 h = (__bf16)src[i];
  dst[(size_t)n * K + k] = __builtin_bit_cast(unsigned short, h);
}

// ---------------------------------------------------------------------------
// WMMA GEMM:  C[M][N] = A[M][K] (fp32, row-major) x Wt ([N][K] bf16) (+bias)(+gelu)
// Block: 256 threads = 8 waves arranged 2(M) x 4(N); each wave computes a
// 16x64 strip (four 16x16 accumulators) -> block tile 32 x 256.
// Requires M%32==0, N%256==0, K%32==0 (true for every call site).
//
// Fragment mapping used here (16-bit, 16x16x32; contiguous K, not the ISA table order):
//   A: lane holds M = mBase + (lane&15), K = kk + (lane>>4)*16 .. +15
//      (same K order as B)  -> four float4 loads + cvt
//   B: lane holds N = nTile + (lane&15), K = kk + (lane>>4)*16 .. +15
//      -> one contiguous 32-byte bf16 load from the pre-transposed weight
//   C/D: vgpr r -> row mBase + (lane>>4)*8 + r, col nTile + (lane&15)
// ---------------------------------------------------------------------------
__launch_bounds__(256)
__global__ void gemm_bf16_wmma(const float* __restrict__ A,
                               const unsigned short* __restrict__ Wt,
                               float* __restrict__ Cmat,
                               const float* __restrict__ bias,
                               int M, int N, int K, int fuseGelu) {
  const int wave = threadIdx.x >> 5;
  const int lane = threadIdx.x & 31;
  const int l16  = lane & 15;
  const int lhi  = lane >> 4;
  const int mBase = blockIdx.y * 32 + (wave >> 2) * 16;
  const int nBase = blockIdx.x * 256 + (wave & 3) * 64;

  const float*          Arow = A  + (size_t)(mBase + l16) * K;
  const unsigned short* B0   = Wt + (size_t)(nBase + l16) * K;
  const size_t tileStride = (size_t)16 * K;   // bf16 elems between 16-col tiles

  floatx8 acc0 = {0.f,0.f,0.f,0.f,0.f,0.f,0.f,0.f};
  floatx8 acc1 = acc0, acc2 = acc0, acc3 = acc0;

  for (int kk = 0; kk < K; kk += 32) {
    const int kA0 = kk + lhi * 16;  // lane half selects K sub-base
    const int kB  = kk + lhi * 16;

    // prefetch one K-step ahead (global_prefetch_b8; speculative, OOB-safe)
    __builtin_prefetch(Arow + kA0 + 32, 0, 1);
    __builtin_prefetch(B0 + kB + 32, 0, 1);

    const float4 a0 = *(const float4*)(Arow + kA0);
    const float4 a1 = *(const float4*)(Arow + kA0 + 4);
    const float4 a2 = *(const float4*)(Arow + kA0 + 8);
    const float4 a3 = *(const float4*)(Arow + kA0 + 12);
    bf16x16 af;
    af[0]  = (__bf16)a0.x; af[1]  = (__bf16)a0.y; af[2]  = (__bf16)a0.z; af[3]  = (__bf16)a0.w;
    af[4]  = (__bf16)a1.x; af[5]  = (__bf16)a1.y; af[6]  = (__bf16)a1.z; af[7]  = (__bf16)a1.w;
    af[8]  = (__bf16)a2.x; af[9]  = (__bf16)a2.y; af[10] = (__bf16)a2.z; af[11] = (__bf16)a2.w;
    af[12] = (__bf16)a3.x; af[13] = (__bf16)a3.y; af[14] = (__bf16)a3.z; af[15] = (__bf16)a3.w;

    const bf16x16 b0 = *(const bf16x16*)(B0 + kB);
    const bf16x16 b1 = *(const bf16x16*)(B0 + tileStride + kB);
    const bf16x16 b2 = *(const bf16x16*)(B0 + 2 * tileStride + kB);
    const bf16x16 b3 = *(const bf16x16*)(B0 + 3 * tileStride + kB);

    acc0 = __builtin_amdgcn_wmma_f32_16x16x32_bf16(false, af, false, b0, (short)0, acc0, false, false);
    acc1 = __builtin_amdgcn_wmma_f32_16x16x32_bf16(false, af, false, b1, (short)0, acc1, false, false);
    acc2 = __builtin_amdgcn_wmma_f32_16x16x32_bf16(false, af, false, b2, (short)0, acc2, false, false);
    acc3 = __builtin_amdgcn_wmma_f32_16x16x32_bf16(false, af, false, b3, (short)0, acc3, false, false);
  }

  const int row0 = mBase + lhi * 8;
#define STORE_TILE(ACC, J)                                                    \
  do {                                                                        \
    const int col = nBase + (J) * 16 + l16;                                   \
    const float bv = bias ? bias[col] : 0.0f;                                 \
    _Pragma("unroll") for (int r = 0; r < 8; ++r) {                           \
      float v = (ACC)[r] + bv;                                                \
      if (fuseGelu) v = gelu_tanh(v);                                         \
      Cmat[(size_t)(row0 + r) * N + col] = v;                                 \
    }                                                                         \
  } while (0)
  STORE_TILE(acc0, 0);
  STORE_TILE(acc1, 1);
  STORE_TILE(acc2, 2);
  STORE_TILE(acc3, 3);
#undef STORE_TILE
}

// ---------------------------------------------------------------------------
// Fused masked-softmax attention. qkv: [nSeq*S][3H] fp32 (q | k | v).
// Grid: (S, NH, nSeq), block 128. mode: 0 = key<lens[seq] (token enc),
// 1 = causal & key<lens[seq] (chunk enc), 2 = causal (decoder).
// ---------------------------------------------------------------------------
__launch_bounds__(128)
__global__ void attn_kernel(const float* __restrict__ qkv, float* __restrict__ out,
                            int S, int mode, const int* __restrict__ lens) {
  const int s = blockIdx.x, h = blockIdx.y, seq = blockIdx.z, tid = threadIdx.x;
  __shared__ float sc[128];
  __shared__ float red[128];
  const size_t stride3H = 3 * Hn;

  const float* qrow = qkv + (size_t)(seq * S + s) * stride3H + h * DH;
  float scj = NEGV;
  if (tid < S) {
    bool valid = true;
    if (mode >= 1) valid = (tid <= s);
    if (mode <= 1) valid = valid && (tid < lens[seq]);
    if (valid) {
      const float* krow = qkv + (size_t)(seq * S + tid) * stride3H + Hn + h * DH;
      float d = 0.f;
#pragma unroll 8
      for (int i = 0; i < DH; ++i) d += qrow[i] * krow[i];
      scj = d * 0.125f;  // 1/sqrt(64)
    }
  }
  sc[tid] = scj;
  red[tid] = scj;
  __syncthreads();
  for (int o = 64; o > 0; o >>= 1) {
    if (tid < o) red[tid] = fmaxf(red[tid], red[tid + o]);
    __syncthreads();
  }
  const float mx = red[0];
  __syncthreads();
  float e = (tid < S) ? __expf(sc[tid] - mx) : 0.0f;
  sc[tid] = e;
  red[tid] = e;
  __syncthreads();
  for (int o = 64; o > 0; o >>= 1) {
    if (tid < o) red[tid] += red[tid + o];
    __syncthreads();
  }
  const float inv = 1.0f / red[0];
  __syncthreads();
  if (tid < DH) {
    float acc = 0.f;
    for (int k = 0; k < S; ++k)
      acc += sc[k] * qkv[(size_t)(seq * S + k) * stride3H + 2 * Hn + h * DH + tid];
    out[(size_t)(seq * S + s) * Hn + h * DH + tid] = acc * inv;
  }
}

// ---------------------------------------------------------------------------
// out[row] = LayerNorm(a[row] (+ b[row])) * lnw[0:H] + lnw[H:2H].  H = 512.
// Block 256 threads, 2 elements per thread. Safe for out == a.
// ---------------------------------------------------------------------------
__launch_bounds__(256)
__global__ void add_ln_kernel(const float* __restrict__ a, const float* __restrict__ b,
                              float* __restrict__ out, const float* __restrict__ lnw) {
  const int row = blockIdx.x, tid = threadIdx.x;
  const float* ar = a + (size_t)row * Hn;
  float v0 = ar[tid], v1 = ar[tid + 256];
  if (b) {
    const float* br = b + (size_t)row * Hn;
    v0 += br[tid];
    v1 += br[tid + 256];
  }
  __shared__ float red[256];
  red[tid] = v0 + v1;
  __syncthreads();
  for (int o = 128; o > 0; o >>= 1) { if (tid < o) red[tid] += red[tid + o]; __syncthreads(); }
  const float mean = red[0] * (1.0f / Hn);
  __syncthreads();
  const float d0 = v0 - mean, d1 = v1 - mean;
  red[tid] = d0 * d0 + d1 * d1;
  __syncthreads();
  for (int o = 128; o > 0; o >>= 1) { if (tid < o) red[tid] += red[tid + o]; __syncthreads(); }
  const float rs = rsqrtf(red[0] * (1.0f / Hn) + 1e-7f);
  out[(size_t)row * Hn + tid]       = d0 * rs * lnw[tid]       + lnw[Hn + tid];
  out[(size_t)row * Hn + tid + 256] = d1 * rs * lnw[tid + 256] + lnw[Hn + tid + 256];
}

// out[row] = LN(emb[ids[row]])
__launch_bounds__(256)
__global__ void embed_ln_kernel(const int* __restrict__ ids, const float* __restrict__ emb,
                                const float* __restrict__ lnw, float* __restrict__ out) {
  const int row = blockIdx.x, tid = threadIdx.x;
  const float* ar = emb + (size_t)ids[row] * Hn;
  float v0 = ar[tid], v1 = ar[tid + 256];
  __shared__ float red[256];
  red[tid] = v0 + v1;
  __syncthreads();
  for (int o = 128; o > 0; o >>= 1) { if (tid < o) red[tid] += red[tid + o]; __syncthreads(); }
  const float mean = red[0] * (1.0f / Hn);
  __syncthreads();
  const float d0 = v0 - mean, d1 = v1 - mean;
  red[tid] = d0 * d0 + d1 * d1;
  __syncthreads();
  for (int o = 128; o > 0; o >>= 1) { if (tid < o) red[tid] += red[tid + o]; __syncthreads(); }
  const float rs = rsqrtf(red[0] * (1.0f / Hn) + 1e-7f);
  out[(size_t)row * Hn + tid]       = d0 * rs * lnw[tid]       + lnw[Hn + tid];
  out[(size_t)row * Hn + tid + 256] = d1 * rs * lnw[tid + 256] + lnw[Hn + tid + 256];
}

// chunk_embs pre-LN: out[i] = tokenUnits[row i*T] + chunk_pos[i % C]
__global__ void gather_chunk_kernel(const float* __restrict__ tokUnits,
                                    const float* __restrict__ chunk_pos,
                                    float* __restrict__ out) {
  const int i = blockIdx.x;  // b*C + c
  const int c = i % Cn;
  const float* src = tokUnits + (size_t)i * Tn * Hn;
  float* o = out + (size_t)i * Hn;
  for (int hh = threadIdx.x; hh < Hn; hh += blockDim.x)
    o[hh] = src[hh] + chunk_pos[(size_t)c * Hn + hh];
}

// chunk_units_sos: out[b*C+0] = sos ; out[b*C+c] = chunkUnits[b*C+c-1]
__global__ void chunk_sos_kernel(const float* __restrict__ chunkUnits,
                                 const float* __restrict__ sos, float* __restrict__ out) {
  const int i = blockIdx.x;
  const int c = i % Cn;
  const float* src = (c == 0) ? sos : chunkUnits + (size_t)(i - 1) * Hn;
  float* o = out + (size_t)i * Hn;
  for (int hh = threadIdx.x; hh < Hn; hh += blockDim.x) o[hh] = src[hh];
}

// dec_in[(b*C+c)*S2 + p] per reference prefix/token conditions
__global__ void dec_in_kernel(const float* __restrict__ chkSos,
                              const float* __restrict__ decTok,
                              const int* __restrict__ num_tokens,
                              const int* __restrict__ num_chunks,
                              float* __restrict__ out) {
  const int idx = blockIdx.x;
  const int p  = idx % S2n;
  const int bc = idx / S2n;
  const int c  = bc % Cn;
  const int b  = bc / Cn;
  const bool cv = c < num_chunks[b];
  const int  n  = num_tokens[bc];
  const float* src = nullptr;
  if (cv) {
    if (p <= c)          src = chkSos + (size_t)(b * Cn + p) * Hn;  // p <= c <= C-1
    else if (p <= c + n) src = decTok + ((size_t)bc * Tn + (p - c - 1)) * Hn;
  }
  float* o = out + (size_t)idx * Hn;
  for (int hh = threadIdx.x; hh < Hn; hh += blockDim.x) o[hh] = src ? src[hh] : 0.0f;
}

// units_re[(bc)*(T-1)+t] = rmask ? units[bc*S2 + c+1+t] : 0
__global__ void gather_units_kernel(const float* __restrict__ units,
                                    const int* __restrict__ num_tokens,
                                    const int* __restrict__ num_chunks,
                                    float* __restrict__ out) {
  const int j  = blockIdx.x;
  const int t  = j % (Tn - 1);
  const int bc = j / (Tn - 1);
  const int c  = bc % Cn;
  const int b  = bc / Cn;
  const bool valid = (t < num_tokens[bc] - 1) && (c < num_chunks[b]);
  const float* src = units + ((size_t)bc * S2n + (c + 1 + t)) * Hn;  // index < 80 always
  float* o = out + (size_t)j * Hn;
  for (int hh = threadIdx.x; hh < Hn; hh += blockDim.x) o[hh] = valid ? src[hh] : 0.0f;
}

// ---------------------------------------------------------------------------
// Host-side orchestration
// ---------------------------------------------------------------------------
static inline void launch_gemm(const float* A, const unsigned short* Wt, float* Cm,
                               const float* bias, int M, int N, int K, int fuseGelu,
                               hipStream_t st) {
  dim3 grid(N / 256, M / 32);
  gemm_bf16_wmma<<<grid, 256, 0, st>>>(A, Wt, Cm, bias, M, N, K, fuseGelu);
}

static inline void launch_wconv(const float* src, unsigned short* dst, int K, int N,
                                hipStream_t st) {
  size_t total = (size_t)K * N;
  int blocks = (int)((total + 255) / 256);
  wconv_kernel<<<blocks, 256, 0, st>>>(src, dst, K, N);
}

struct EncPtrs {
  const unsigned short *wqkvT, *woT, *w1T, *w2T;
  const float *ln1, *ln2;
};

static void run_encoder(float* X, int nSeq, int S, const EncPtrs& w, int mode,
                        const int* lens, float* QKV, float* ATT, float* Y, float* Hb,
                        hipStream_t st) {
  const int rows = nSeq * S;
  for (int l = 0; l < Ln; ++l) {
    launch_gemm(X, w.wqkvT + (size_t)l * 3 * Hn * Hn, QKV, nullptr, rows, 3 * Hn, Hn, 0, st);
    attn_kernel<<<dim3(S, NH, nSeq), 128, 0, st>>>(QKV, ATT, S, mode, lens);
    launch_gemm(ATT, w.woT + (size_t)l * Hn * Hn, Y, nullptr, rows, Hn, Hn, 0, st);
    add_ln_kernel<<<rows, 256, 0, st>>>(X, Y, X, w.ln1 + (size_t)l * 2 * Hn);
    launch_gemm(X, w.w1T + (size_t)l * Fn * Hn, Hb, nullptr, rows, Fn, Hn, 1, st);
    launch_gemm(Hb, w.w2T + (size_t)l * Hn * Fn, Y, nullptr, rows, Hn, Fn, 0, st);
    add_ln_kernel<<<rows, 256, 0, st>>>(X, Y, X, w.ln2 + (size_t)l * 2 * Hn);
  }
}

extern "C" void kernel_launch(void* const* d_in, const int* in_sizes, int n_in,
                              void* d_out, int out_size, void* d_ws, size_t ws_size,
                              hipStream_t stream) {
  (void)in_sizes; (void)n_in; (void)out_size; (void)ws_size;

  const int*   token_ids    = (const int*)d_in[0];
  const int*   num_chunks   = (const int*)d_in[1];
  const int*   num_tokens   = (const int*)d_in[2];
  const float* tok_emb      = (const float*)d_in[3];
  const float* tok_emb_ln   = (const float*)d_in[4];
  const float* chunk_pos    = (const float*)d_in[5];
  const float* chunk_emb_ln = (const float*)d_in[6];
  const float* sos          = (const float*)d_in[7];
  const float* dec_emb      = (const float*)d_in[8];
  const float* dec_emb_ln   = (const float*)d_in[9];
  const float* cls_dense    = (const float*)d_in[10];
  const float* cls_ln       = (const float*)d_in[11];
  const float* cls_proj     = (const float*)d_in[12];
  const float* cls_b        = (const float*)d_in[13];
  // per-encoder weights: [wqkv, wo, ln1, w1, w2, ln2] for tok(14), chk(20), dec(26)
  const float* encw[3][6];
  for (int e = 0; e < 3; ++e)
    for (int i = 0; i < 6; ++i) encw[e][i] = (const float*)d_in[14 + e * 6 + i];

  // ---- carve workspace ----------------------------------------------------
  char* p = (char*)d_ws;
  auto carve = [&](size_t bytes) -> void* {
    void* r = (void*)p;
    p += (bytes + 255) & ~(size_t)255;
    return r;
  };

  const size_t szQKV = (size_t)Ln * 3 * Hn * Hn;  // bf16 elements
  const size_t szWO  = (size_t)Ln * Hn * Hn;
  const size_t szW1  = (size_t)Ln * Fn * Hn;
  const size_t szW2  = (size_t)Ln * Hn * Fn;
  unsigned short* wT[3][4];
  for (int e = 0; e < 3; ++e) {
    wT[e][0] = (unsigned short*)carve(szQKV * 2);
    wT[e][1] = (unsigned short*)carve(szWO * 2);
    wT[e][2] = (unsigned short*)carve(szW1 * 2);
    wT[e][3] = (unsigned short*)carve(szW2 * 2);
  }
  unsigned short* clsDT = (unsigned short*)carve((size_t)Hn * Hn * 2);
  unsigned short* clsPT = (unsigned short*)carve((size_t)Vn * Hn * 2);

  const int maxRows = Bn * Cn * S2n;  // 2560
  float* X      = (float*)carve((size_t)maxRows * Hn * 4);
  float* BIG    = (float*)carve((size_t)maxRows * Fn * 4);  // shared QKV / FFN h
  float* ATT    = (float*)carve((size_t)maxRows * Hn * 4);
  float* Y      = (float*)carve((size_t)maxRows * Hn * 4);
  float* decTok = (float*)carve((size_t)(Bn * Cn * Tn) * Hn * 4);
  float* chkX   = (float*)carve((size_t)(Bn * Cn) * Hn * 4);
  float* chkSos = (float*)carve((size_t)(Bn * Cn) * Hn * 4);
  float* uRe    = (float*)carve((size_t)(Bn * Cn * (Tn - 1)) * Hn * 4);
  float* Y2     = (float*)carve((size_t)(Bn * Cn * (Tn - 1)) * Hn * 4);

  // ---- weight conversion (fp32 [K][N] -> bf16 [N][K]) ---------------------
  for (int e = 0; e < 3; ++e) {
    for (int l = 0; l < Ln; ++l) {
      launch_wconv(encw[e][0] + (size_t)l * Hn * 3 * Hn, wT[e][0] + (size_t)l * 3 * Hn * Hn, Hn, 3 * Hn, stream);
      launch_wconv(encw[e][1] + (size_t)l * Hn * Hn,     wT[e][1] + (size_t)l * Hn * Hn,     Hn, Hn,     stream);
      launch_wconv(encw[e][3] + (size_t)l * Hn * Fn,     wT[e][2] + (size_t)l * Fn * Hn,     Hn, Fn,     stream);
      launch_wconv(encw[e][4] + (size_t)l * Fn * Hn,     wT[e][3] + (size_t)l * Hn * Fn,     Fn, Hn,     stream);
    }
  }
  launch_wconv(cls_dense, clsDT, Hn, Hn, stream);
  launch_wconv(cls_proj,  clsPT, Hn, Vn, stream);

  // ---- token encoder ------------------------------------------------------
  embed_ln_kernel<<<Bn * Cn * Tn, 256, 0, stream>>>(token_ids, tok_emb, tok_emb_ln, X);
  EncPtrs tokE{wT[0][0], wT[0][1], wT[0][2], wT[0][3], encw[0][2], encw[0][5]};
  run_encoder(X, Bn * Cn, Tn, tokE, /*mode=*/0, num_tokens, BIG, ATT, Y, BIG, stream);

  // ---- chunk encoder ------------------------------------------------------
  gather_chunk_kernel<<<Bn * Cn, 256, 0, stream>>>(X, chunk_pos, chkX);
  add_ln_kernel<<<Bn * Cn, 256, 0, stream>>>(chkX, nullptr, chkX, chunk_emb_ln);
  EncPtrs chkE{wT[1][0], wT[1][1], wT[1][2], wT[1][3], encw[1][2], encw[1][5]};
  run_encoder(chkX, Bn, Cn, chkE, /*mode=*/1, num_chunks, BIG, ATT, Y, BIG, stream);
  chunk_sos_kernel<<<Bn * Cn, 256, 0, stream>>>(chkX, sos, chkSos);

  // ---- decoder ------------------------------------------------------------
  embed_ln_kernel<<<Bn * Cn * Tn, 256, 0, stream>>>(token_ids, dec_emb, dec_emb_ln, decTok);
  dec_in_kernel<<<Bn * Cn * S2n, 256, 0, stream>>>(chkSos, decTok, num_tokens, num_chunks, X);
  EncPtrs decE{wT[2][0], wT[2][1], wT[2][2], wT[2][3], encw[2][2], encw[2][5]};
  run_encoder(X, Bn * Cn, S2n, decE, /*mode=*/2, nullptr, BIG, ATT, Y, BIG, stream);

  // ---- classifier head ----------------------------------------------------
  gather_units_kernel<<<Bn * Cn * (Tn - 1), 256, 0, stream>>>(X, num_tokens, num_chunks, uRe);
  launch_gemm(uRe, clsDT, Y2, nullptr, Bn * Cn * (Tn - 1), Hn, Hn, /*gelu=*/1, stream);
  add_ln_kernel<<<Bn * Cn * (Tn - 1), 256, 0, stream>>>(Y2, nullptr, Y2, cls_ln);
  launch_gemm(Y2, clsPT, (float*)d_out, cls_b, Bn * Cn * (Tn - 1), Vn, Hn, /*gelu=*/0, stream);
}
